// SCAM_43465069036272
// MI455X (gfx1250) — matrix operand order repaired and checked by reading, not compiled, not run on hardware
//
#include <hip/hip_runtime.h>

#define Bq 4
#define Cq 64
#define Hq 96
#define Wq 384
#define WP 392          // LDS stride for [C][W] f32 buffers (2*WP%64==16 -> halves hit disjoint banks)
#define VST 65          // LDS stride for [W][C] V buffers
#define PPB 424         // bf16 P-panel stride (elements); 424*2B rows keep 16B alignment
#define MTB 424         // bf16 B-chunk row stride (elements)
#define NTHREADS 256

typedef __attribute__((ext_vector_type(2))) float v2f;
typedef __attribute__((ext_vector_type(8))) float v8f;
typedef __bf16 bf16_t;
typedef __attribute__((ext_vector_type(8)))  __bf16 v8bf;
typedef __attribute__((ext_vector_type(16))) __bf16 v16bf;

static __device__ __forceinline__ v8f wmma4(v2f a, v2f b, v8f c) {
    // V_WMMA_F32_16X16X4_F32 : D = A(16x4) x B(4x16) + C
    return __builtin_amdgcn_wmma_f32_16x16x4_f32(false, a, false, b, (short)0, c, false, false);
}
static __device__ __forceinline__ v8f wmma32bf(v16bf a, v16bf b, v8f c) {
    // V_WMMA_F32_16X16X32_BF16 : D = A(16x32) x B(32x16) + C
    return __builtin_amdgcn_wmma_f32_16x16x32_bf16(false, a, false, b, (short)0, c, false, false);
}

static __device__ __forceinline__ float wave_sum(float v) {
    for (int o = 16; o > 0; o >>= 1) v += __shfl_xor(v, o, 32);
    return v;
}
static __device__ __forceinline__ float wave_max(float v) {
    for (int o = 16; o > 0; o >>= 1) v = fmaxf(v, __shfl_xor(v, o, 32));
    return v;
}
static __device__ __forceinline__ float block_sum(float v, float* red) {
    int t = threadIdx.x;
    red[t] = v; __syncthreads();
    for (int s = 128; s > 0; s >>= 1) { if (t < s) red[t] += red[t + s]; __syncthreads(); }
    float r = red[0]; __syncthreads();
    return r;
}

// ---------------------------------------------------------------------------
// K1: per-(b,h): LayerNorm + Q projections (f32 WMMA), attention A = Q_l^T Q_r
//     * scale (f32 WMMA), softmax stats (row/col max+sum), validity masks.
// ---------------------------------------------------------------------------
__global__ void __launch_bounds__(NTHREADS)
k_attn(const float* __restrict__ xl, const float* __restrict__ xr,
       const float* __restrict__ nlw, const float* __restrict__ nlb,
       const float* __restrict__ nrw, const float* __restrict__ nrb,
       const float* __restrict__ w1l, const float* __restrict__ b1l,
       const float* __restrict__ w1r, const float* __restrict__ b1r,
       float* __restrict__ Ag,
       float* __restrict__ g_rm, float* __restrict__ g_rs,
       float* __restrict__ g_cm, float* __restrict__ g_cs,
       float* __restrict__ g_mkL, float* __restrict__ g_mkR)
{
    extern __shared__ float sm[];
    float* Ql  = sm;                    // [Cq][WP]
    float* Qr  = Ql + Cq * WP;
    float* Xb  = Qr + Cq * WP;          // staging / xn
    float* muL = Xb + Cq * WP;          // [Wq]
    float* rsd = muL + Wq;
    float* rmS = rsd + Wq;
    float* cmS = rmS + Wq;
    float* rsS = cmS + Wq;
    float* csS = rsS + Wq;

    const int row = blockIdx.x, b = row / Hq, h = row % Hq;
    const int tid = threadIdx.x, wave = tid >> 5, lane = tid & 31;
    const int half = lane >> 4, lm = lane & 15;

    auto do_side = [&](const float* xg, const float* nw, const float* nb,
                       const float* w1, const float* b1, float* Qdst) {
        for (int t = tid; t < Cq * Wq; t += NTHREADS) {
            int c = t / Wq, w = t - c * Wq;
            Xb[c * WP + w] = xg[((size_t)(b * Cq + c) * Hq + h) * Wq + w];
        }
        __syncthreads();
        for (int w = tid; w < Wq; w += NTHREADS) {
            float s = 0.f, s2 = 0.f;
            for (int c = 0; c < Cq; ++c) { float v = Xb[c * WP + w]; s += v; s2 += v * v; }
            float mu = s * (1.f / Cq);
            float var = s2 * (1.f / Cq) - mu * mu;
            muL[w] = mu; rsd[w] = rsqrtf(var + 1e-6f);
        }
        __syncthreads();
        for (int t = tid; t < Cq * Wq; t += NTHREADS) {
            int c = t / Wq, w = t - c * Wq;
            float v = Xb[c * WP + w];
            Xb[c * WP + w] = (v - muL[w]) * rsd[w] * nw[c] + nb[c];
        }
        __syncthreads();
        // Q = w1 @ xn + b1   (tiles: 4 o-chunks x 24 w-chunks)
        for (int tI = wave; tI < 96; tI += 8) {
            int o0 = (tI / 24) * 16, w0 = (tI % 24) * 16;
            v8f acc = {};
            for (int cc = 0; cc < Cq; cc += 4) {
                v2f a, bb;
                a[0] = w1[(o0 + lm) * Cq + cc + 2 * half];
                a[1] = w1[(o0 + lm) * Cq + cc + 2 * half + 1];
                bb[0] = Xb[(cc + 2 * half) * WP + w0 + lm];
                bb[1] = Xb[(cc + 2 * half + 1) * WP + w0 + lm];
                acc = wmma4(a, bb, acc);
            }
            for (int r = 0; r < 8; ++r) {
                int o = o0 + r + 8 * half;
                Qdst[o * WP + w0 + lm] = acc[r] + b1[o];
            }
        }
        __syncthreads();
    };

    do_side(xl, nlw, nlb, w1l, b1l, Ql);
    do_side(xr, nrw, nrb, w1r, b1r, Qr);

    float* Arow = Ag + (size_t)row * Wq * Wq;
    // A = Q_l^T Q_r * scale
    for (int tI = wave; tI < 576; tI += 8) {
        int w0 = (tI / 24) * 16, v0 = (tI % 24) * 16;
        v8f acc = {};
        for (int cc = 0; cc < Cq; cc += 4) {
            v2f a, bb;
            a[0]  = Ql[(cc + 2 * half) * WP + w0 + lm];
            a[1]  = Ql[(cc + 2 * half + 1) * WP + w0 + lm];
            bb[0] = Qr[(cc + 2 * half) * WP + v0 + lm];
            bb[1] = Qr[(cc + 2 * half + 1) * WP + v0 + lm];
            acc = wmma4(a, bb, acc);
        }
        for (int r = 0; r < 8; ++r) {
            int w = w0 + r + 8 * half;
            Arow[(size_t)w * Wq + v0 + lm] = acc[r] * 0.125f;
        }
    }
    __threadfence();
    __syncthreads();

    // row max / col max
    for (int w = wave; w < Wq; w += 8) {
        float mx = -1e30f;
        for (int v = lane; v < Wq; v += 32) mx = fmaxf(mx, Arow[(size_t)w * Wq + v]);
        mx = wave_max(mx);
        if (lane == 0) rmS[w] = mx;
    }
    for (int g = wave; g < 12; g += 8) {
        int v = g * 32 + lane;
        float mx = -1e30f;
        for (int w = 0; w < Wq; ++w) mx = fmaxf(mx, Arow[(size_t)w * Wq + v]);
        cmS[v] = mx;
    }
    __syncthreads();
    // row sum / col sum
    for (int w = wave; w < Wq; w += 8) {
        float s = 0.f, mx = rmS[w];
        for (int v = lane; v < Wq; v += 32) s += __expf(Arow[(size_t)w * Wq + v] - mx);
        s = wave_sum(s);
        if (lane == 0) rsS[w] = s;
    }
    for (int g = wave; g < 12; g += 8) {
        int v = g * 32 + lane;
        float s = 0.f, mx = cmS[v];
        for (int w = 0; w < Wq; ++w) s += __expf(Arow[(size_t)w * Wq + v] - mx);
        csS[v] = s;
    }
    __syncthreads();
    // masks: V_l2r[w] = sum_v Ml[v,w] > 0.1 ; V_r2l[v] = sum_w P[w,v] > 0.1
    for (int w = wave; w < Wq; w += 8) {
        float s = 0.f;
        for (int v = lane; v < Wq; v += 32)
            s += __expf(Arow[(size_t)w * Wq + v] - cmS[v]) / csS[v];
        s = wave_sum(s);
        if (lane == 0) g_mkL[row * Wq + w] = (s > 0.1f) ? 1.f : 0.f;
    }
    for (int g = wave; g < 12; g += 8) {
        int v = g * 32 + lane;
        float s = 0.f;
        for (int w = 0; w < Wq; ++w)
            s += __expf(Arow[(size_t)w * Wq + v] - rmS[w]) / rsS[w];
        g_mkR[row * Wq + v] = (s > 0.1f) ? 1.f : 0.f;
    }
    for (int w = tid; w < Wq; w += NTHREADS) {
        g_rm[row * Wq + w] = rmS[w]; g_rs[row * Wq + w] = rsS[w];
        g_cm[row * Wq + w] = cmS[w]; g_cs[row * Wq + w] = csS[w];
    }
}

// ---------------------------------------------------------------------------
// K2: per-(b,h): V = w2@x+b (f32 WMMA), out1 = x_l + beta*(M_r2l @ V_r),
//     out2 = x_r + gamma*(M_l2r @ V_l) (f32 WMMA, P built on the fly), photo loss.
// ---------------------------------------------------------------------------
__global__ void __launch_bounds__(NTHREADS)
k_out(const float* __restrict__ xl, const float* __restrict__ xr,
      const float* __restrict__ lrl, const float* __restrict__ lrr,
      const float* __restrict__ w2l, const float* __restrict__ b2l,
      const float* __restrict__ w2r, const float* __restrict__ b2r,
      const float* __restrict__ beta, const float* __restrict__ gamma,
      const float* __restrict__ Ag,
      const float* __restrict__ g_rm, const float* __restrict__ g_rs,
      const float* __restrict__ g_cm, const float* __restrict__ g_cs,
      const float* __restrict__ g_mkL, const float* __restrict__ g_mkR,
      float* __restrict__ out1, float* __restrict__ out2,
      float* __restrict__ pPhoto)
{
    extern __shared__ float sm[];
    float* Vl   = sm;                   // [Wq][VST]
    float* Vr   = Vl + Wq * VST;
    float* Xb   = Vr + Wq * VST;        // [Cq][WP]
    float* rmL  = Xb + Cq * WP;
    float* rsiL = rmL + Wq;
    float* cmL  = rsiL + Wq;
    float* csiL = cmL + Wq;
    float* mkLL = csiL + Wq;
    float* mkRL = mkLL + Wq;
    float* lrLs = mkRL + Wq;            // [3][Wq]
    float* lrRs = lrLs + 3 * Wq;
    float* red  = lrRs + 3 * Wq;        // [256]

    const int row = blockIdx.x, b = row / Hq, h = row % Hq;
    const int tid = threadIdx.x, wave = tid >> 5, lane = tid & 31;
    const int half = lane >> 4, lm = lane & 15;
    const float* Arow = Ag + (size_t)row * Wq * Wq;

    for (int w = tid; w < Wq; w += NTHREADS) {
        rmL[w] = g_rm[row * Wq + w]; rsiL[w] = 1.f / g_rs[row * Wq + w];
        cmL[w] = g_cm[row * Wq + w]; csiL[w] = 1.f / g_cs[row * Wq + w];
        mkLL[w] = g_mkL[row * Wq + w]; mkRL[w] = g_mkR[row * Wq + w];
    }
    for (int t = tid; t < 3 * Wq; t += NTHREADS) {
        int c = t / Wq, w = t - c * Wq;
        lrLs[t] = lrl[((size_t)(b * 3 + c) * Hq + h) * Wq + w];
        lrRs[t] = lrr[((size_t)(b * 3 + c) * Hq + h) * Wq + w];
    }

    auto make_v = [&](const float* xg, const float* w2, const float* b2, float* Vd) {
        for (int t = tid; t < Cq * Wq; t += NTHREADS) {
            int c = t / Wq, w = t - c * Wq;
            Xb[c * WP + w] = xg[((size_t)(b * Cq + c) * Hq + h) * Wq + w];
        }
        __syncthreads();
        for (int tI = wave; tI < 96; tI += 8) {
            int o0 = (tI / 24) * 16, w0 = (tI % 24) * 16;
            v8f acc = {};
            for (int cc = 0; cc < Cq; cc += 4) {
                v2f a, bb;
                a[0] = w2[(o0 + lm) * Cq + cc + 2 * half];
                a[1] = w2[(o0 + lm) * Cq + cc + 2 * half + 1];
                bb[0] = Xb[(cc + 2 * half) * WP + w0 + lm];
                bb[1] = Xb[(cc + 2 * half + 1) * WP + w0 + lm];
                acc = wmma4(a, bb, acc);
            }
            for (int r = 0; r < 8; ++r) {
                int o = o0 + r + 8 * half;
                Vd[(w0 + lm) * VST + o] = acc[r] + b2[o];
            }
        }
        __syncthreads();
    };
    make_v(xl, w2l, b2l, Vl);
    make_v(xr, w2r, b2r, Vr);

    // out1 = x_l + beta * (P @ V_r)
    for (int tI = wave; tI < 96; tI += 8) {
        int w0 = (tI / 4) * 16, c0 = (tI % 4) * 16;
        int wI = w0 + lm;
        float rmv = rmL[wI], rsv = rsiL[wI];
        v8f acc = {};
        for (int vv = 0; vv < Wq; vv += 4) {
            v2f a, bb;
            const float2 av = *reinterpret_cast<const float2*>(Arow + (size_t)wI * Wq + vv + 2 * half);
            a[0] = __expf(av.x - rmv) * rsv;
            a[1] = __expf(av.y - rmv) * rsv;
            bb[0] = Vr[(vv + 2 * half) * VST + c0 + lm];
            bb[1] = Vr[(vv + 2 * half + 1) * VST + c0 + lm];
            acc = wmma4(a, bb, acc);
        }
        for (int r = 0; r < 8; ++r) {
            int w = w0 + r + 8 * half, c = c0 + lm;
            size_t idx = ((size_t)(b * Cq + c) * Hq + h) * Wq + w;
            out1[idx] = xl[idx] + beta[c] * acc[r];
        }
    }
    // out2 = x_r + gamma * (Ml @ V_l)
    for (int tI = wave; tI < 96; tI += 8) {
        int i0 = (tI / 4) * 16, c0 = (tI % 4) * 16;
        int iI = i0 + lm;
        float cmv = cmL[iI], csv = csiL[iI];
        v8f acc = {};
        for (int vv = 0; vv < Wq; vv += 4) {
            v2f a, bb;
            a[0] = __expf(Arow[(size_t)(vv + 2 * half) * Wq + iI] - cmv) * csv;
            a[1] = __expf(Arow[(size_t)(vv + 2 * half + 1) * Wq + iI] - cmv) * csv;
            bb[0] = Vl[(vv + 2 * half) * VST + c0 + lm];
            bb[1] = Vl[(vv + 2 * half + 1) * VST + c0 + lm];
            acc = wmma4(a, bb, acc);
        }
        for (int r = 0; r < 8; ++r) {
            int i = i0 + r + 8 * half, c = c0 + lm;
            size_t idx = ((size_t)(b * Cq + c) * Hq + h) * Wq + i;
            out2[idx] = xr[idx] + gamma[c] * acc[r];
        }
    }

    // photo loss
    float tP = 0.f;
    for (int w = wave; w < Wq; w += 8) {
        float rmv = rmL[w], rsv = rsiL[w];
        float s0 = 0.f, s1 = 0.f, s2 = 0.f;
        for (int v = lane; v < Wq; v += 32) {
            float p = __expf(Arow[(size_t)w * Wq + v] - rmv) * rsv;
            s0 += p * lrRs[v]; s1 += p * lrRs[Wq + v]; s2 += p * lrRs[2 * Wq + v];
        }
        s0 = wave_sum(s0); s1 = wave_sum(s1); s2 = wave_sum(s2);
        if (lane == 0) {
            float m = mkLL[w];
            tP += m * (fabsf(lrLs[w] - s0) + fabsf(lrLs[Wq + w] - s1) + fabsf(lrLs[2 * Wq + w] - s2));
        }
    }
    for (int i = tid; i < Wq; i += NTHREADS) {
        float cmv = cmL[i], csv = csiL[i];
        float s0 = 0.f, s1 = 0.f, s2 = 0.f;
        for (int j = 0; j < Wq; ++j) {
            float e = __expf(Arow[(size_t)j * Wq + i] - cmv) * csv;
            s0 += e * lrLs[j]; s1 += e * lrLs[Wq + j]; s2 += e * lrLs[2 * Wq + j];
        }
        float m = mkRL[i];
        tP += m * (fabsf(lrRs[i] - s0) + fabsf(lrRs[Wq + i] - s1) + fabsf(lrRs[2 * Wq + i] - s2));
    }
    float s = block_sum(tP, red);
    if (tid == 0) pPhoto[row] = s;
}

// ---------------------------------------------------------------------------
// K3: per-(b,h): cycle loss via bf16 WMMA (16x16x32) on LDS-staged probability
//     panels, plus loss_w (shifted-diagonal smoothness). Loss-only -> bf16 safe.
// ---------------------------------------------------------------------------
__global__ void __launch_bounds__(NTHREADS)
k_cycle(const float* __restrict__ Ag,
        const float* __restrict__ g_rm, const float* __restrict__ g_rs,
        const float* __restrict__ g_cm, const float* __restrict__ g_cs,
        const float* __restrict__ g_mkL, const float* __restrict__ g_mkR,
        float* __restrict__ pCyc, float* __restrict__ pW)
{
    extern __shared__ float sm[];
    float* rmL  = sm;
    float* rsiL = rmL + Wq;
    float* cmL  = rsiL + Wq;
    float* csiL = cmL + Wq;
    float* mkLL = csiL + Wq;
    float* mkRL = mkLL + Wq;
    float* red  = mkRL + Wq;                 // [256]
    bf16_t* Pp  = (bf16_t*)(red + 256);      // [16][PPB]  A-operand panel
    bf16_t* McT = Pp + 16 * PPB;             // [128][MTB] B-operand chunk, K-contiguous

    const int row = blockIdx.x;
    const int tid = threadIdx.x, wave = tid >> 5, lane = tid & 31;
    const int half = lane >> 4, lm = lane & 15;
    const float* Arow = Ag + (size_t)row * Wq * Wq;

    for (int w = tid; w < Wq; w += NTHREADS) {
        rmL[w] = g_rm[row * Wq + w]; rsiL[w] = 1.f / g_rs[row * Wq + w];
        cmL[w] = g_cm[row * Wq + w]; csiL[w] = 1.f / g_cs[row * Wq + w];
        mkLL[w] = g_mkL[row * Wq + w]; mkRL[w] = g_mkR[row * Wq + w];
    }
    __syncthreads();

    float tW = 0.f, tC = 0.f;
    // loss_w : |M[w,v]-M[w+1,v+1]| for both softmax matrices (f32)
    for (int w = wave; w < Wq - 1; w += 8) {
        for (int v = lane; v < Wq - 1; v += 32) {
            float a1 = Arow[(size_t)w * Wq + v];
            float a2 = Arow[(size_t)(w + 1) * Wq + v + 1];
            float p1 = __expf(a1 - rmL[w]) * rsiL[w];
            float p2 = __expf(a2 - rmL[w + 1]) * rsiL[w + 1];
            float q1 = __expf(a1 - cmL[v]) * csiL[v];
            float q2 = __expf(a2 - cmL[v + 1]) * csiL[v + 1];
            tW += fabsf(p1 - p2) + fabsf(q1 - q2);
        }
    }

    // cycle loss: ph=0 -> M_lrl = P @ Ml ; ph=1 -> M_rlr = Ml @ P
    for (int ph = 0; ph < 2; ++ph) {
        for (int uc = 0; uc < 3; ++uc) {       // output-column chunks of 128
            __syncthreads();
            if (ph == 0) {
                // B chunk: McT[ul][k=v] = Ml[v][u] = exp(A[u,v]-cm[v])/cs[v]
                for (int t = tid; t < 128 * Wq; t += NTHREADS) {
                    int ul = t / Wq, v = t - ul * Wq, u = uc * 128 + ul;
                    McT[ul * MTB + v] = (bf16_t)(__expf(Arow[(size_t)u * Wq + v] - cmL[v]) * csiL[v]);
                }
            } else {
                // B chunk: McT[ul][k=w] = P[w][u] = exp(A[w,u]-rm[w])/rs[w]
                for (int t = tid; t < 128 * Wq; t += NTHREADS) {
                    int ul = t & 127, w = t >> 7, u = uc * 128 + ul;
                    McT[ul * MTB + w] = (bf16_t)(__expf(Arow[(size_t)w * Wq + u] - rmL[w]) * rsiL[w]);
                }
            }
            __syncthreads();
            for (int r0 = 0; r0 < 24; ++r0) {  // output-row 16-chunks
                int m0 = r0 * 16;
                __syncthreads();
                if (ph == 0) {
                    for (int t = tid; t < 16 * Wq; t += NTHREADS) {
                        int m = t / Wq, v = t - m * Wq;
                        Pp[m * PPB + v] = (bf16_t)(__expf(Arow[(size_t)(m0 + m) * Wq + v] - rmL[m0 + m]) * rsiL[m0 + m]);
                    }
                } else {
                    for (int t = tid; t < 16 * Wq; t += NTHREADS) {
                        int m = t & 15, w = t >> 4;
                        Pp[m * PPB + w] = (bf16_t)(__expf(Arow[(size_t)w * Wq + m0 + m] - cmL[m0 + m]) * csiL[m0 + m]);
                    }
                }
                __syncthreads();
                const int u0 = wave * 16;
                v8f acc = {};
                for (int vv = 0; vv < Wq; vv += 32) {
                    // 16-bit A layout: lanes 0-15 K=0..7,16..23 ; lanes 16-31 K=8..15,24..31
                    v8bf alo = *(const v8bf*)(Pp + lm * PPB + vv + 8 * half);
                    v8bf ahi = *(const v8bf*)(Pp + lm * PPB + vv + 16 + 8 * half);
                    // B layout: lanes 0-15 K=0..7,16..23 ; lanes 16-31 K=8..15,24..31 (K-contiguous in McT)
                    v8bf blo = *(const v8bf*)(McT + (u0 + lm) * MTB + vv + 8 * half);
                    v8bf bhi = *(const v8bf*)(McT + (u0 + lm) * MTB + vv + 16 + 8 * half);
                    v16bf a = __builtin_shufflevector(alo, ahi, 0, 1, 2, 3, 4, 5, 6, 7,
                                                      8, 9, 10, 11, 12, 13, 14, 15);
                    v16bf bb = __builtin_shufflevector(blo, bhi, 0, 1, 2, 3, 4, 5, 6, 7,
                                                       8, 9, 10, 11, 12, 13, 14, 15);
                    acc = wmma32bf(a, bb, acc);
                }
                int uG = uc * 128 + u0 + lm;
                float msk = (ph == 0) ? mkLL[uG] : mkRL[uG];
                for (int r = 0; r < 8; ++r) {
                    int mo = m0 + r + 8 * half;
                    tC += msk * fabsf(acc[r] - ((mo == uG) ? 1.f : 0.f));
                }
            }
        }
    }
    float sC = block_sum(tC, red);
    float sW = block_sum(tW, red);
    if (tid == 0) { pCyc[row] = sC; pW[row] = sW; }
}

// ---------------------------------------------------------------------------
// K4: adjacent-h smoothness loss
// ---------------------------------------------------------------------------
__global__ void __launch_bounds__(NTHREADS)
k_lossh(const float* __restrict__ Ag,
        const float* __restrict__ g_rm, const float* __restrict__ g_rs,
        const float* __restrict__ g_cm, const float* __restrict__ g_cs,
        float* __restrict__ pH)
{
    extern __shared__ float sm[];
    float* rm0 = sm;            float* rsi0 = rm0 + Wq;
    float* cm0 = rsi0 + Wq;     float* csi0 = cm0 + Wq;
    float* rm1 = csi0 + Wq;     float* rsi1 = rm1 + Wq;
    float* cm1 = rsi1 + Wq;     float* csi1 = cm1 + Wq;
    float* red = csi1 + Wq;

    const int blk = blockIdx.x;
    const int b = blk / (Hq - 1), hh = blk % (Hq - 1);
    const int r0 = b * Hq + hh, r1 = r0 + 1;
    const int tid = threadIdx.x, wave = tid >> 5, lane = tid & 31;
    const float* A0 = Ag + (size_t)r0 * Wq * Wq;
    const float* A1 = Ag + (size_t)r1 * Wq * Wq;

    for (int w = tid; w < Wq; w += NTHREADS) {
        rm0[w] = g_rm[r0 * Wq + w]; rsi0[w] = 1.f / g_rs[r0 * Wq + w];
        cm0[w] = g_cm[r0 * Wq + w]; csi0[w] = 1.f / g_cs[r0 * Wq + w];
        rm1[w] = g_rm[r1 * Wq + w]; rsi1[w] = 1.f / g_rs[r1 * Wq + w];
        cm1[w] = g_cm[r1 * Wq + w]; csi1[w] = 1.f / g_cs[r1 * Wq + w];
    }
    __syncthreads();

    float t = 0.f;
    for (int w = wave; w < Wq; w += 8) {
        for (int v = lane; v < Wq; v += 32) {
            float a0 = A0[(size_t)w * Wq + v];
            float a1 = A1[(size_t)w * Wq + v];
            t += fabsf(__expf(a0 - rm0[w]) * rsi0[w] - __expf(a1 - rm1[w]) * rsi1[w]);
            t += fabsf(__expf(a0 - cm0[v]) * csi0[v] - __expf(a1 - cm1[v]) * csi1[v]);
        }
    }
    float s = block_sum(t, red);
    if (tid == 0) pH[blk] = s;
}

// ---------------------------------------------------------------------------
// K5: passthrough copies of LR images
// ---------------------------------------------------------------------------
__global__ void __launch_bounds__(NTHREADS)
k_copy(const float* __restrict__ lrl, const float* __restrict__ lrr,
       float* __restrict__ out3, float* __restrict__ out4, int n)
{
    int i = blockIdx.x * NTHREADS + threadIdx.x;
    if (i < n) { out3[i] = lrl[i]; out4[i] = lrr[i]; }
}

// ---------------------------------------------------------------------------
// K6: deterministic loss finalize
// ---------------------------------------------------------------------------
__global__ void __launch_bounds__(NTHREADS)
k_final(const float* __restrict__ pPhoto, const float* __restrict__ pCyc,
        const float* __restrict__ pW, const float* __restrict__ pH,
        const float* __restrict__ loss_in, float* __restrict__ out_loss)
{
    extern __shared__ float sm[];
    const int tid = threadIdx.x;
    float sp = 0.f, sc = 0.f, sw = 0.f, sh = 0.f;
    for (int i = tid; i < Bq * Hq; i += NTHREADS) { sp += pPhoto[i]; sc += pCyc[i]; sw += pW[i]; }
    for (int i = tid; i < Bq * (Hq - 1); i += NTHREADS) sh += pH[i];
    sp = block_sum(sp, sm); sc = block_sum(sc, sm);
    sw = block_sum(sw, sm); sh = block_sum(sh, sm);
    if (tid == 0) {
        const float Np = (float)(Bq * 3 * Hq * Wq);
        const float Nw = (float)((size_t)Bq * Hq * (Wq - 1) * (Wq - 1));
        const float Nh = (float)((size_t)Bq * (Hq - 1) * Wq * Wq);
        const float Nc = (float)((size_t)Bq * Hq * Wq * Wq);
        float l = loss_in[0] + 0.0025f * (sp / Np + 0.1f * (sw / Nw + sh / Nh) + sc / Nc);
        out_loss[0] = l;
    }
}

// ---------------------------------------------------------------------------
extern "C" void kernel_launch(void* const* d_in, const int* in_sizes, int n_in,
                              void* d_out, int out_size, void* d_ws, size_t ws_size,
                              hipStream_t stream)
{
    const float* xl   = (const float*)d_in[0];
    const float* xr   = (const float*)d_in[1];
    const float* lrl  = (const float*)d_in[2];
    const float* lrr  = (const float*)d_in[3];
    const float* lin  = (const float*)d_in[4];
    const float* nlw  = (const float*)d_in[5];
    const float* nlb  = (const float*)d_in[6];
    const float* nrw  = (const float*)d_in[7];
    const float* nrb  = (const float*)d_in[8];
    const float* w1l  = (const float*)d_in[9];
    const float* b1l  = (const float*)d_in[10];
    const float* w1r  = (const float*)d_in[11];
    const float* b1r  = (const float*)d_in[12];
    const float* w2l  = (const float*)d_in[13];
    const float* b2l  = (const float*)d_in[14];
    const float* w2r  = (const float*)d_in[15];
    const float* b2r  = (const float*)d_in[16];
    const float* beta = (const float*)d_in[17];
    const float* gmma = (const float*)d_in[18];

    float* ws = (float*)d_ws;
    const size_t NA = (size_t)Bq * Hq * Wq * Wq;   // 56,623,104 floats
    const size_t NR = (size_t)Bq * Hq * Wq;        // 147,456 floats
    float* Ag  = ws;
    float* rm  = Ag + NA;
    float* rs  = rm + NR;
    float* cm  = rs + NR;
    float* cs  = cm + NR;
    float* mkL = cs + NR;
    float* mkR = mkL + NR;
    float* pPhoto = mkR + NR;
    float* pCyc   = pPhoto + Bq * Hq;
    float* pW     = pCyc + Bq * Hq;
    float* pH     = pW + Bq * Hq;

    float* out1 = (float*)d_out;
    float* out2 = out1 + (size_t)Bq * Cq * Hq * Wq;
    float* out3 = out2 + (size_t)Bq * Cq * Hq * Wq;
    float* out4 = out3 + (size_t)Bq * 3 * Hq * Wq;
    float* outL = out4 + (size_t)Bq * 3 * Hq * Wq;

    const size_t SH1 = (size_t)(3 * Cq * WP + 6 * Wq) * sizeof(float);
    const size_t SH2 = (size_t)(2 * Wq * VST + Cq * WP + 12 * Wq + 256) * sizeof(float);
    const size_t SH3 = (size_t)(6 * Wq + 256) * sizeof(float)
                     + (size_t)(16 * PPB + 128 * MTB) * sizeof(bf16_t);
    const size_t SH4 = (size_t)(8 * Wq + 256) * sizeof(float);
    const size_t SH6 = (size_t)256 * sizeof(float);

    dim3 blk(NTHREADS);
    hipLaunchKernelGGL(k_attn, dim3(Bq * Hq), blk, SH1, stream,
                       xl, xr, nlw, nlb, nrw, nrb, w1l, b1l, w1r, b1r,
                       Ag, rm, rs, cm, cs, mkL, mkR);
    hipLaunchKernelGGL(k_out, dim3(Bq * Hq), blk, SH2, stream,
                       xl, xr, lrl, lrr, w2l, b2l, w2r, b2r, beta, gmma,
                       Ag, rm, rs, cm, cs, mkL, mkR, out1, out2, pPhoto);
    hipLaunchKernelGGL(k_cycle, dim3(Bq * Hq), blk, SH3, stream,
                       Ag, rm, rs, cm, cs, mkL, mkR, pCyc, pW);
    hipLaunchKernelGGL(k_lossh, dim3(Bq * (Hq - 1)), blk, SH4, stream,
                       Ag, rm, rs, cm, cs, pH);
    const int nLR = Bq * 3 * Hq * Wq;
    hipLaunchKernelGGL(k_copy, dim3((nLR + NTHREADS - 1) / NTHREADS), blk, 0, stream,
                       lrl, lrr, out3, out4, nLR);
    hipLaunchKernelGGL(k_final, dim3(1), blk, SH6, stream,
                       pPhoto, pCyc, pW, pH, lin, outL);
}
